// VQVariationalAutoencoder_6219112645202
// MI455X (gfx1250) — matrix operand order repaired and checked by reading, not compiled, not run on hardware
//
#include <hip/hip_runtime.h>
#include <hip/hip_bf16.h>

// ---------------------------------------------------------------------------
// VQ-VAE forward for MI455X (gfx1250).
// All five GEMMs via v_wmma_f32_16x16x32_bf16 with fp32 accumulate.
// Operands pre-converted to padded bf16 (weights pre-transposed) so the GEMM
// inner loop is guard-free 16B global->LDS staging, double-buffered and using
// the async-to-LDS ASYNCcnt path (global_load_async_to_lds_b128).
// ---------------------------------------------------------------------------

#define BATCH    16384
#define IN_DIM   700
#define H_ENC    2048
#define LATENT   512
#define H_DEC    2048
#define OUT_DIM  700
#define NUM_EMB  1024
#define KPAD_X   704      // IN_DIM padded to multiple of 32
#define NPAD_OUT 768      // OUT_DIM padded to multiple of 128

typedef __bf16 bf16_t;
typedef __attribute__((ext_vector_type(16))) __bf16 v16bf;
typedef __attribute__((ext_vector_type(8)))  __bf16 v8bf;
typedef __attribute__((ext_vector_type(8)))  float  v8f;
typedef __attribute__((ext_vector_type(4)))  unsigned int u32x4;
typedef __attribute__((ext_vector_type(4)))  int v4i_t;

__device__ __forceinline__ bf16_t f2bf(float f) {
    union { float f; unsigned u; } in; in.f = f;
    unsigned u = in.u + 0x7FFFu + ((in.u >> 16) & 1u);   // round-to-nearest-even
    union { unsigned short s; bf16_t b; } out;
    out.s = (unsigned short)(u >> 16);
    return out.b;
}

// 16-byte global -> LDS copy. Prefer the CDNA5 async-to-LDS instruction
// (ASYNCcnt-tracked DMA, frees the wave); fall back to load+ds_store.
__device__ __forceinline__ void cp16_g2l(const void* g, void* l) {
#if defined(__gfx1250__) && __has_builtin(__builtin_amdgcn_global_load_async_to_lds_b128)
    __builtin_amdgcn_global_load_async_to_lds_b128(
        (__attribute__((address_space(1))) v4i_t*)(void*)(size_t)g,
        (__attribute__((address_space(3))) v4i_t*)l, 0, 0);
#else
    *(u32x4*)l = *(const u32x4*)g;
#endif
}
__device__ __forceinline__ void cp16_wait() {
#if defined(__gfx1250__) && __has_builtin(__builtin_amdgcn_global_load_async_to_lds_b128) && \
    __has_builtin(__builtin_amdgcn_s_wait_asynccnt)
    __builtin_amdgcn_s_wait_asynccnt(0);
#endif
}

// ---------------------------------------------------------------------------
// Tiled WMMA GEMM:  C[M,N] = act(A[M,Kpad] @ Bt[N,Kpad]^T + bias)
// A, Bt: bf16, zero-padded, 16B-aligned rows. Block tile 128x128, BK=32,
// double-buffered LDS (one barrier per K-step). 256 threads = 8 wave32s;
// each wave owns a 32x64 sub-tile => 2x4 accumulators of 16x16 f32.
// OUTMODE: 0 = f32 only, 1 = bf16 only, 2 = both. ACT: 1 = leaky_relu(0.2).
// ---------------------------------------------------------------------------
template <int ACT, int OUTMODE>
__global__ __launch_bounds__(256)
void wmma_gemm_kernel(const bf16_t* __restrict__ A, const bf16_t* __restrict__ Bt,
                      const float* __restrict__ bias,
                      float* __restrict__ Cf, bf16_t* __restrict__ Cb,
                      int Kpad, int N)
{
    // row stride 56 bf16 = 112 B: multiple of 16 B (aligned b128) and a 28-bank
    // stride -> conflict-free across 16 consecutive rows.
    __shared__ bf16_t lA[2][128][56];   // [buf][m][k]
    __shared__ bf16_t lB[2][128][56];   // [buf][n][k]

    const int tid     = threadIdx.x;
    const int lane    = tid & 31;
    const int wid     = tid >> 5;
    const int warpM   = wid >> 1;                  // 0..3  (32-row slabs)
    const int warpN   = wid & 1;                   // 0..1  (64-col slabs)
    const int rowBase = blockIdx.y * 128;
    const int colBase = blockIdx.x * 128;          // grid.x covers padded N
    const int l15     = lane & 15;
    const int halfSel = (lane >> 4) & 1;

    v8f acc[2][4];
#pragma unroll
    for (int i = 0; i < 2; ++i)
#pragma unroll
        for (int j = 0; j < 4; ++j)
            acc[i][j] = (v8f)(0.0f);

    // Stage one 128x32 tile of A and Bt into buffer b. Guard-free: buffers are
    // padded. Each thread moves 2x16B for A and 2x16B for B (all b128 traffic).
    auto stage = [&](int b, int kt) {
#pragma unroll
        for (int i = 0; i < 2; ++i) {
            const int id = tid + i * 256;          // 512 chunks of 8 bf16
            const int r  = id >> 2;
            const int c  = (id & 3) * 8;
            cp16_g2l(A + (size_t)(rowBase + r) * Kpad + kt + c, &lA[b][r][c]);
        }
#pragma unroll
        for (int i = 0; i < 2; ++i) {
            const int id = tid + i * 256;
            const int r  = id >> 2;
            const int c  = (id & 3) * 8;
            cp16_g2l(Bt + (size_t)(colBase + r) * Kpad + kt + c, &lB[b][r][c]);
        }
    };

    stage(0, 0);
    cp16_wait();
    __syncthreads();

    int buf = 0;
    for (int kt = 0; kt < Kpad; kt += 32) {
        if (kt + 32 < Kpad) stage(buf ^ 1, kt + 32);   // prefetch next tile

        // A 16x32 frag: lanes 0-15 hold K {0..7,16..23}; lanes 16-31 {8..15,24..31}
        v16bf afrag[2];
#pragma unroll
        for (int tm = 0; tm < 2; ++tm) {
            const int r  = warpM * 32 + tm * 16 + l15;
            const int kb = halfSel * 8;
            union { v16bf v; v8bf h[2]; } u;
            u.h[0] = *(const v8bf*)&lA[buf][r][kb];
            u.h[1] = *(const v8bf*)&lA[buf][r][kb + 16];
            afrag[tm] = u.v;
        }
        // B 32x16 frag: lanes 0-15 hold K {0..7,16..23} of col N=lane; lanes 16-31 {8..15,24..31}
        v16bf bfrag[4];
#pragma unroll
        for (int tn = 0; tn < 4; ++tn) {
            const int c  = warpN * 64 + tn * 16 + l15;
            const int kb = halfSel * 8;
            union { v16bf v; v8bf h[2]; } u;
            u.h[0] = *(const v8bf*)&lB[buf][c][kb];
            u.h[1] = *(const v8bf*)&lB[buf][c][kb + 16];
            bfrag[tn] = u.v;
        }

#pragma unroll
        for (int tm = 0; tm < 2; ++tm)
#pragma unroll
            for (int tn = 0; tn < 4; ++tn)
                acc[tm][tn] = __builtin_amdgcn_wmma_f32_16x16x32_bf16(
                    false, afrag[tm], false, bfrag[tn],
                    (short)0, acc[tm][tn], false, false);

        cp16_wait();        // our prefetch landed
        __syncthreads();    // everyone done reading buf & done staging buf^1
        buf ^= 1;
    }

    // Epilogue: bias + optional leaky-relu; C/D layout: VGPR i -> M = i + 8*halfSel
#pragma unroll
    for (int tm = 0; tm < 2; ++tm) {
#pragma unroll
        for (int tn = 0; tn < 4; ++tn) {
            const int gcol = colBase + warpN * 64 + tn * 16 + l15;
            if (gcol >= N) continue;
            const float bv = bias ? bias[gcol] : 0.0f;
#pragma unroll
            for (int i = 0; i < 8; ++i) {
                const int grow = rowBase + warpM * 32 + tm * 16 + halfSel * 8 + i;
                float v = acc[tm][tn][i] + bv;
                if (ACT) v = v >= 0.0f ? v : 0.2f * v;
                const size_t o = (size_t)grow * N + gcol;
                if (OUTMODE == 0 || OUTMODE == 2) Cf[o] = v;
                if (OUTMODE == 1 || OUTMODE == 2) Cb[o] = f2bf(v);
            }
        }
    }
}

// ---- prep: x [B,700] f32 -> xbf [B,704] bf16 zero-padded ---------------------
__global__ __launch_bounds__(256)
void pad_x_kernel(const float* __restrict__ x, bf16_t* __restrict__ xbf)
{
    const size_t t = (size_t)blockIdx.x * 256 + threadIdx.x;
    if (t >= (size_t)BATCH * KPAD_X) return;
    const int row = (int)(t / KPAD_X);
    const int k   = (int)(t % KPAD_X);
    xbf[t] = (k < IN_DIM) ? f2bf(x[(size_t)row * IN_DIM + k]) : f2bf(0.0f);
}

// ---- prep: W [K,N] f32 -> Wt [Npad,Kpad] bf16, transposed + zero-padded ------
__global__ __launch_bounds__(256)
void transpose_w_kernel(const float* __restrict__ W, bf16_t* __restrict__ Wt,
                        int K, int N, int Kpad, int Npad)
{
    const size_t t = (size_t)blockIdx.x * 256 + threadIdx.x;
    if (t >= (size_t)Npad * Kpad) return;
    const int n = (int)(t / Kpad);
    const int k = (int)(t % Kpad);
    Wt[t] = (n < N && k < K) ? f2bf(W[(size_t)k * N + n]) : f2bf(0.0f);
}

// ---- codebook column norms ---------------------------------------------------
__global__ __launch_bounds__(256)
void enorm_kernel(const float* __restrict__ emb, float* __restrict__ enorm)
{
    const int n = blockIdx.x * blockDim.x + threadIdx.x;
    if (n < NUM_EMB) {
        float s = 0.0f;
        for (int k = 0; k < LATENT; ++k) {
            const float e = emb[(size_t)k * NUM_EMB + n];
            s += e * e;
        }
        enorm[n] = s;
    }
}

// ---- argmin over codebook: one wave32 per batch row --------------------------
// dists differ from the reference by row-constant ||z||^2 -> same argmin.
__global__ __launch_bounds__(256)
void argmin_kernel(const float* __restrict__ s, const float* __restrict__ enorm,
                   int* __restrict__ idx)
{
    const int lane = threadIdx.x & 31;
    const int row  = blockIdx.x * 8 + (threadIdx.x >> 5);
    const float* sr = s + (size_t)row * NUM_EMB;
    float best = 3.402823466e38f;
    int   bi   = 0;
    for (int n = lane; n < NUM_EMB; n += 32) {
        const float v = enorm[n] - 2.0f * sr[n];
        if (v < best) { best = v; bi = n; }
    }
#pragma unroll
    for (int off = 16; off > 0; off >>= 1) {
        const float ov = __shfl_xor(best, off, 32);
        const int   oi = __shfl_xor(bi,   off, 32);
        if (ov < best || (ov == best && oi < bi)) { best = ov; bi = oi; }
    }
    if (lane == 0) idx[row] = bi;
}

__global__ void zero_loss_kernel(float* loss) { *loss = 0.0f; }

// ---- gather quantized (as bf16 for decoder GEMM) + VQ loss -------------------
// vq_loss = (BETA + 1) * mean((q - z)^2) = 1.25 * mean(...); q_st == q.
__global__ __launch_bounds__(256)
void gather_loss_kernel(const float* __restrict__ emb, const float* __restrict__ z,
                        const int* __restrict__ idx, bf16_t* __restrict__ qbf,
                        float* __restrict__ loss)
{
    __shared__ float red[8];
    const size_t t = (size_t)blockIdx.x * 256 + threadIdx.x;   // over BATCH*LATENT
    const int b = (int)(t >> 9);            // LATENT = 512
    const int l = (int)(t & 511);
    const int code = idx[b];
    const float qv = emb[(size_t)l * NUM_EMB + code];          // emb 2MB -> L2 resident
    const float d  = qv - z[t];
    qbf[t] = f2bf(qv);

    float v = d * d;
#pragma unroll
    for (int off = 16; off > 0; off >>= 1) v += __shfl_xor(v, off, 32);
    const int lane = threadIdx.x & 31;
    const int wid  = threadIdx.x >> 5;
    if (lane == 0) red[wid] = v;
    __syncthreads();
    if (threadIdx.x == 0) {
        float s = 0.0f;
#pragma unroll
        for (int w = 0; w < 8; ++w) s += red[w];
        const float scale = 1.25f / (float)((size_t)BATCH * LATENT);
        atomicAdd(loss, s * scale);
    }
}

// ---------------------------------------------------------------------------
extern "C" void kernel_launch(void* const* d_in, const int* in_sizes, int n_in,
                              void* d_out, int out_size, void* d_ws, size_t ws_size,
                              hipStream_t stream)
{
    (void)in_sizes; (void)n_in; (void)out_size; (void)ws_size;
    const float* x   = (const float*)d_in[0];
    const float* W1  = (const float*)d_in[1];
    const float* b1  = (const float*)d_in[2];
    const float* W2  = (const float*)d_in[3];
    const float* b2  = (const float*)d_in[4];
    const float* emb = (const float*)d_in[5];
    const float* W3  = (const float*)d_in[6];
    const float* b3  = (const float*)d_in[7];
    const float* W4  = (const float*)d_in[8];
    const float* b4  = (const float*)d_in[9];

    float* out  = (float*)d_out;
    float* loss = out + (size_t)BATCH * OUT_DIM;

    // Workspace layout (~163 MB); region0 (64MB) serves h_bf -> s(f32) -> h2_bf.
    char* p = (char*)d_ws;
    bf16_t* xbf  = (bf16_t*)p; p += (size_t)BATCH * KPAD_X * 2;       // 23.1 MB
    bf16_t* W1t  = (bf16_t*)p; p += (size_t)H_ENC * KPAD_X * 2;       //  2.9 MB
    bf16_t* W2t  = (bf16_t*)p; p += (size_t)LATENT * H_ENC * 2;       //  2.0 MB
    bf16_t* embT = (bf16_t*)p; p += (size_t)NUM_EMB * LATENT * 2;     //  1.0 MB
    bf16_t* W3t  = (bf16_t*)p; p += (size_t)H_DEC * LATENT * 2;       //  2.0 MB
    bf16_t* W4t  = (bf16_t*)p; p += (size_t)NPAD_OUT * H_DEC * 2;     //  3.1 MB
    char*   r0   = p;          p += (size_t)BATCH * H_ENC * 2;        // 64 MB region0
    bf16_t* hbf  = (bf16_t*)r0;                 // encoder hidden (bf16)
    float*  s    = (float*)r0;                  // 16384x1024 f32 = 64 MB (h dead)
    bf16_t* h2bf = (bf16_t*)r0;                 // decoder hidden (s dead)
    float*  zf   = (float*)p;  p += (size_t)BATCH * LATENT * 4;       // 32 MB
    bf16_t* zbf  = (bf16_t*)p; p += (size_t)BATCH * LATENT * 2;       // 16 MB
    bf16_t* qbf  = (bf16_t*)p; p += (size_t)BATCH * LATENT * 2;       // 16 MB
    int*    idx  = (int*)p;    p += (size_t)BATCH * 4;
    float*  enorm= (float*)p;

    const dim3 blk(256);
    const int mTiles = BATCH / 128;
    auto g1 = [](size_t n) { return dim3((unsigned)((n + 255) / 256)); };

    // ---- prep: bf16 conversion, padding, weight transposes ----
    pad_x_kernel<<<g1((size_t)BATCH * KPAD_X), blk, 0, stream>>>(x, xbf);
    transpose_w_kernel<<<g1((size_t)H_ENC * KPAD_X), blk, 0, stream>>>(W1, W1t, IN_DIM, H_ENC, KPAD_X, H_ENC);
    transpose_w_kernel<<<g1((size_t)LATENT * H_ENC), blk, 0, stream>>>(W2, W2t, H_ENC, LATENT, H_ENC, LATENT);
    transpose_w_kernel<<<g1((size_t)NUM_EMB * LATENT), blk, 0, stream>>>(emb, embT, LATENT, NUM_EMB, LATENT, NUM_EMB);
    transpose_w_kernel<<<g1((size_t)H_DEC * LATENT), blk, 0, stream>>>(W3, W3t, LATENT, H_DEC, LATENT, H_DEC);
    transpose_w_kernel<<<g1((size_t)NPAD_OUT * H_DEC), blk, 0, stream>>>(W4, W4t, H_DEC, OUT_DIM, H_DEC, NPAD_OUT);
    enorm_kernel<<<dim3(NUM_EMB / 256), blk, 0, stream>>>(emb, enorm);

    // 1) h = leaky(x @ W1 + b1) -> bf16            [16384,704]x[704,2048]
    wmma_gemm_kernel<1, 1><<<dim3(H_ENC / 128, mTiles), blk, 0, stream>>>(
        xbf, W1t, b1, nullptr, hbf, KPAD_X, H_ENC);
    // 2) z = h @ W2 + b2 -> f32 + bf16             [16384,2048]x[2048,512]
    wmma_gemm_kernel<0, 2><<<dim3(LATENT / 128, mTiles), blk, 0, stream>>>(
        hbf, W2t, b2, zf, zbf, H_ENC, LATENT);
    // 3) s = z @ emb -> f32 (reuses region0)       [16384,512]x[512,1024]
    wmma_gemm_kernel<0, 0><<<dim3(NUM_EMB / 128, mTiles), blk, 0, stream>>>(
        zbf, embT, nullptr, s, nullptr, LATENT, NUM_EMB);
    // 4) idx = argmin_n(enorm[n] - 2 s[b,n])
    argmin_kernel<<<dim3(BATCH / 8), blk, 0, stream>>>(s, enorm, idx);
    // 5) loss = 0; gather q (bf16) + 1.25*mean((q-z)^2)
    zero_loss_kernel<<<1, 1, 0, stream>>>(loss);
    gather_loss_kernel<<<g1((size_t)BATCH * LATENT), blk, 0, stream>>>(emb, zf, idx, qbf, loss);
    // 6) h2 = leaky(q @ W3 + b3) -> bf16 (region0) [16384,512]x[512,2048]
    wmma_gemm_kernel<1, 1><<<dim3(H_DEC / 128, mTiles), blk, 0, stream>>>(
        qbf, W3t, b3, nullptr, h2bf, LATENT, H_DEC);
    // 7) recon = h2 @ W4 + b4 -> d_out             [16384,2048]x[2048,768->700]
    wmma_gemm_kernel<0, 0><<<dim3(NPAD_OUT / 128, mTiles), blk, 0, stream>>>(
        h2bf, W4t, b4, out, nullptr, H_DEC, OUT_DIM);
}
